// Net_60052232733175
// MI455X (gfx1250) — matrix operand order repaired and checked by reading, not compiled, not run on hardware
//
#include <hip/hip_runtime.h>
#include <hip/hip_bf16.h>

#define HDIM 128
#define NGRAPHS 256
#define NCLASSES 10
#define LDS_STRIDE 136   // 128 + 8 bf16 pad -> 272B row stride, spreads LDS banks

typedef __attribute__((ext_vector_type(16))) __bf16          v16bf;
typedef __attribute__((ext_vector_type(8)))  float           v8f;
typedef __attribute__((ext_vector_type(8)))  unsigned short  us8;

union BF16x16 { v16bf v; unsigned short u[16]; us8 h[2]; };

__device__ __forceinline__ unsigned short f32_to_bf16(float f) {
    unsigned u = __float_as_uint(f);
    unsigned r = u + 0x7FFFu + ((u >> 16) & 1u);   // round-to-nearest-even
    return (unsigned short)(r >> 16);
}

// ---------------------------------------------------------------------------
// zero fill
// ---------------------------------------------------------------------------
__global__ void zero_kernel(float* __restrict__ p, size_t n) {
    size_t t = (size_t)blockIdx.x * blockDim.x + threadIdx.x;
    if (t < n) p[t] = 0.0f;
}

// ---------------------------------------------------------------------------
// f32 -> bf16 conversion (weights, once per launch)
// ---------------------------------------------------------------------------
__global__ void cvt_bf16_kernel(const float* __restrict__ in,
                                unsigned short* __restrict__ out, int n) {
    int t = blockIdx.x * blockDim.x + threadIdx.x;
    if (t < n) out[t] = f32_to_bf16(in[t]);
}

// ---------------------------------------------------------------------------
// scatter-add aggregation: agg[dst] += x[src]   (one wave-lane = 4 channels)
// ---------------------------------------------------------------------------
__global__ void scatter_add_kernel(const float* __restrict__ xin,
                                   const int* __restrict__ src,
                                   const int* __restrict__ dst,
                                   float* __restrict__ agg, int n_edges) {
    size_t t = (size_t)blockIdx.x * blockDim.x + threadIdx.x;
    int e  = (int)(t >> 5);
    int c4 = (int)(t & 31) * 4;
    if (e >= n_edges) return;
    int s = src[e];
    int d = dst[e];
    float4 v = *(const float4*)(xin + (size_t)s * HDIM + c4);
    float* o = agg + (size_t)d * HDIM + c4;
    atomicAdd(o + 0, v.x);
    atomicAdd(o + 1, v.y);
    atomicAdd(o + 2, v.z);
    atomicAdd(o + 3, v.w);
}

// ---------------------------------------------------------------------------
// GraphConv GEMM epilogue:
//   out = relu(agg @ Wr^T + xin @ Wl^T + b) [+ residual]
// Block = 256 threads = 8 waves; block covers 16 nodes x 128 channels.
// A tiles staged once per block into LDS as bf16; each wave then computes a
// 16x16 tile with 8x v_wmma_f32_16x16x32_bf16 (4 K-tiles x {Wr,Wl}).
// A fragments are two contiguous 16B ds loads per tile (CDNA5 wave32 layout:
// lane elements 0..7 <-> K=khalf..khalf+7, 8..15 <-> K=16+khalf..+7).
// ---------------------------------------------------------------------------
__global__ void __launch_bounds__(256)
conv_gemm_kernel(const float* __restrict__ agg, const float* __restrict__ xin,
                 const unsigned short* __restrict__ WrB,
                 const unsigned short* __restrict__ WlB,
                 const float* __restrict__ bias, const float* __restrict__ residual,
                 float* __restrict__ xout, int n_nodes) {
    __shared__ __align__(16) unsigned short lds_agg[16 * LDS_STRIDE];
    __shared__ __align__(16) unsigned short lds_x[16 * LDS_STRIDE];

    const int tid  = threadIdx.x;
    const int lane = tid & 31;
    const int wave = tid >> 5;                 // 0..7 -> 16-channel tile
    const int node_base = blockIdx.x * 16;
    const int h_base = wave * 16;

    // ---- cooperative stage: 16x128 f32 (agg,x) -> bf16 LDS; 8 elems/thread
    {
        const int row = tid >> 4;              // 0..15
        const int col = (tid & 15) * 8;        // 0,8,...,120
        const float* pa = agg + (size_t)(node_base + row) * HDIM + col;
        const float* px = xin + (size_t)(node_base + row) * HDIM + col;
        const float4 a0 = *(const float4*)(pa);
        const float4 a1 = *(const float4*)(pa + 4);
        const float4 x0 = *(const float4*)(px);
        const float4 x1 = *(const float4*)(px + 4);
        us8 va, vx;
        va[0] = f32_to_bf16(a0.x); va[1] = f32_to_bf16(a0.y);
        va[2] = f32_to_bf16(a0.z); va[3] = f32_to_bf16(a0.w);
        va[4] = f32_to_bf16(a1.x); va[5] = f32_to_bf16(a1.y);
        va[6] = f32_to_bf16(a1.z); va[7] = f32_to_bf16(a1.w);
        vx[0] = f32_to_bf16(x0.x); vx[1] = f32_to_bf16(x0.y);
        vx[2] = f32_to_bf16(x0.z); vx[3] = f32_to_bf16(x0.w);
        vx[4] = f32_to_bf16(x1.x); vx[5] = f32_to_bf16(x1.y);
        vx[6] = f32_to_bf16(x1.z); vx[7] = f32_to_bf16(x1.w);
        *(us8*)(&lds_agg[row * LDS_STRIDE + col]) = va;
        *(us8*)(&lds_x[row * LDS_STRIDE + col])   = vx;
        // prefetch next block's rows while we compute (global_prefetch_b8)
        if (node_base + 16 + row < n_nodes) {
            __builtin_prefetch(agg + (size_t)(node_base + 16 + row) * HDIM + col, 0, 3);
            __builtin_prefetch(xin + (size_t)(node_base + 16 + row) * HDIM + col, 0, 3);
        }
    }
    __syncthreads();

    const int m     = lane & 15;               // A row (node within tile)
    const int khalf = (lane >> 4) << 3;        // A: 0 or 8
    const int nn    = lane & 15;               // B column (out channel)
    const int koff  = (lane >> 4) << 3;        // B: 0 or 8

    v8f acc = {};

#pragma unroll
    for (int kt = 0; kt < 4; ++kt) {
        const int kb = kt * 32;
        BF16x16 a_agg, a_x, b_r, b_l;
        // A fragments: two contiguous 16B runs from LDS
        a_agg.h[0] = *(const us8*)(&lds_agg[m * LDS_STRIDE + kb + khalf]);
        a_agg.h[1] = *(const us8*)(&lds_agg[m * LDS_STRIDE + kb + 16 + khalf]);
        a_x.h[0]   = *(const us8*)(&lds_x[m * LDS_STRIDE + kb + khalf]);
        a_x.h[1]   = *(const us8*)(&lds_x[m * LDS_STRIDE + kb + 16 + khalf]);
        // B fragments: B[k][n] = W[h_base+n][k], two 8-bf16 runs at kb+koff and kb+16+koff
        const unsigned short* wr = WrB + (size_t)(h_base + nn) * HDIM + kb + koff;
        const unsigned short* wl = WlB + (size_t)(h_base + nn) * HDIM + kb + koff;
        b_r.h[0] = *(const us8*)(wr);
        b_r.h[1] = *(const us8*)(wr + 16);
        b_l.h[0] = *(const us8*)(wl);
        b_l.h[1] = *(const us8*)(wl + 16);

        acc = __builtin_amdgcn_wmma_f32_16x16x32_bf16(false, a_agg.v, false, b_r.v,
                                                      (short)0, acc, false, false);
        acc = __builtin_amdgcn_wmma_f32_16x16x32_bf16(false, a_x.v, false, b_l.v,
                                                      (short)0, acc, false, false);
    }

    // epilogue: bias + relu (+ residual); C/D layout: vgpr r -> M = r + 8*(lane>>4)
    const int h = h_base + (lane & 15);
    const float bv = bias[h];
#pragma unroll
    for (int r = 0; r < 8; ++r) {
        const int node = node_base + r + ((lane >> 4) << 3);
        float v = acc[r] + bv;
        v = v > 0.0f ? v : 0.0f;
        if (residual) v += residual[(size_t)node * HDIM + h];
        xout[(size_t)node * HDIM + h] = v;
    }
}

// ---------------------------------------------------------------------------
// pooling: per (node, channel) -> atomic max (bits, values >= 0) + atomic sum
// ---------------------------------------------------------------------------
__global__ void pool_kernel(const float* __restrict__ x4, const int* __restrict__ batch,
                            unsigned* __restrict__ gm_bits, float* __restrict__ gs,
                            float* __restrict__ cnt, int n_nodes) {
    size_t t = (size_t)blockIdx.x * blockDim.x + threadIdx.x;
    int node = (int)(t >> 7);
    int c    = (int)(t & 127);
    if (node >= n_nodes) return;
    int g = batch[node];
    float v = x4[(size_t)node * HDIM + c];
    atomicMax(&gm_bits[(size_t)g * HDIM + c], __float_as_uint(v)); // v >= 0 (post-relu)
    atomicAdd(&gs[(size_t)g * HDIM + c], v);
    if (c == 0) atomicAdd(&cnt[g], 1.0f);
}

__global__ void pool_finish_kernel(const unsigned* __restrict__ gm_bits,
                                   const float* __restrict__ gs,
                                   const float* __restrict__ cnt,
                                   float* __restrict__ h1) {
    int t = blockIdx.x * blockDim.x + threadIdx.x;
    int g = t >> 7;
    int c = t & 127;
    if (g >= NGRAPHS) return;
    float ct = cnt[g];
    ct = ct > 1.0f ? ct : 1.0f;
    h1[(size_t)g * 256 + c]       = __uint_as_float(gm_bits[(size_t)g * HDIM + c]);
    h1[(size_t)g * 256 + 128 + c] = gs[(size_t)g * HDIM + c] / ct;
}

// ---------------------------------------------------------------------------
// MLP head: out[g,o] = (relu)(b[o] + sum_d in[g,d] * W[o,d])
// ---------------------------------------------------------------------------
__global__ void linear_kernel(const float* __restrict__ in, const float* __restrict__ W,
                              const float* __restrict__ b, float* __restrict__ out,
                              int G, int Din, int Dout, int do_relu) {
    int t = blockIdx.x * blockDim.x + threadIdx.x;
    int g = t / Dout;
    int o = t % Dout;
    if (g >= G) return;
    float s = b[o];
    const float* row = in + (size_t)g * Din;
    const float* wr  = W + (size_t)o * Din;
    for (int d = 0; d < Din; ++d) s += row[d] * wr[d];
    if (do_relu) s = fmaxf(s, 0.0f);
    out[(size_t)g * Dout + o] = s;
}

__global__ void logsoftmax_kernel(const float* __restrict__ in, float* __restrict__ out,
                                  int G, int C) {
    int g = blockIdx.x * blockDim.x + threadIdx.x;
    if (g >= G) return;
    const float* row = in + (size_t)g * C;
    float mx = row[0];
    for (int c = 1; c < C; ++c) mx = fmaxf(mx, row[c]);
    float s = 0.0f;
    for (int c = 0; c < C; ++c) s += __expf(row[c] - mx);
    float lse = mx + __logf(s);
    for (int c = 0; c < C; ++c) out[(size_t)g * C + c] = row[c] - lse;
}

// ---------------------------------------------------------------------------
extern "C" void kernel_launch(void* const* d_in, const int* in_sizes, int n_in,
                              void* d_out, int out_size, void* d_ws, size_t ws_size,
                              hipStream_t stream) {
    const float* x     = (const float*)d_in[0];
    const int*   ei    = (const int*)d_in[1];
    const int*   batch = (const int*)d_in[2];
    const float* w1r = (const float*)d_in[3];
    const float* w1l = (const float*)d_in[4];
    const float* b1  = (const float*)d_in[5];
    const float* w2r = (const float*)d_in[6];
    const float* w2l = (const float*)d_in[7];
    const float* b2  = (const float*)d_in[8];
    const float* w3r = (const float*)d_in[9];
    const float* w3l = (const float*)d_in[10];
    const float* b3  = (const float*)d_in[11];
    const float* lw1 = (const float*)d_in[12];
    const float* lb1 = (const float*)d_in[13];
    const float* lw2 = (const float*)d_in[14];
    const float* lb2 = (const float*)d_in[15];
    const float* lw3 = (const float*)d_in[16];
    const float* lb3 = (const float*)d_in[17];

    const int n_nodes = in_sizes[0] / HDIM;     // 50000
    const int n_edges = in_sizes[1] / 2;        // 800000
    const int* src = ei;
    const int* dst = ei + n_edges;

    // ---- workspace layout ----
    // front: 6 bf16 weight matrices (128x128 each) = 98304 ushorts = 49152 floats
    const int WN = HDIM * HDIM;                 // 16384
    unsigned short* wb   = (unsigned short*)d_ws;
    unsigned short* w1rB = wb + 0 * WN;
    unsigned short* w1lB = wb + 1 * WN;
    unsigned short* w2rB = wb + 2 * WN;
    unsigned short* w2lB = wb + 3 * WN;
    unsigned short* w3rB = wb + 4 * WN;
    unsigned short* w3lB = wb + 5 * WN;

    float* fbase = (float*)d_ws + (6 * WN) / 2; // 49152 floats in
    const size_t NH = (size_t)n_nodes * HDIM;
    float* agg = fbase;
    float* t1  = agg + NH;
    float* t2  = t1 + NH;
    float* t3  = t2 + NH;
    float* t4  = t3 + NH;
    float* gm  = t4 + NH;                       // used as uint bits, 256*128
    float* gs  = gm + (size_t)NGRAPHS * HDIM;
    float* cnt = gs + (size_t)NGRAPHS * HDIM;
    float* h1  = cnt + NGRAPHS;                 // 256*256
    float* h2  = h1 + (size_t)NGRAPHS * 256;    // 256*128
    float* h3  = h2 + (size_t)NGRAPHS * HDIM;   // 256*64
    float* h4  = h3 + (size_t)NGRAPHS * 64;     // 256*10

    const int TB = 256;
    const int zero_blocks    = (int)((NH + TB - 1) / TB);
    const int scatter_blocks = (int)(((size_t)n_edges * 32 + TB - 1) / TB);
    const int conv_blocks    = n_nodes / 16;    // 3125 (N divisible by 16)
    const int pool_blocks    = (int)(((size_t)n_nodes * HDIM + TB - 1) / TB);
    const int cvt_blocks     = (WN + TB - 1) / TB;

    // ---- weight conversion (once per launch) ----
    cvt_bf16_kernel<<<cvt_blocks, TB, 0, stream>>>(w1r, w1rB, WN);
    cvt_bf16_kernel<<<cvt_blocks, TB, 0, stream>>>(w1l, w1lB, WN);
    cvt_bf16_kernel<<<cvt_blocks, TB, 0, stream>>>(w2r, w2rB, WN);
    cvt_bf16_kernel<<<cvt_blocks, TB, 0, stream>>>(w2l, w2lB, WN);
    cvt_bf16_kernel<<<cvt_blocks, TB, 0, stream>>>(w3r, w3rB, WN);
    cvt_bf16_kernel<<<cvt_blocks, TB, 0, stream>>>(w3l, w3lB, WN);

    // ---- layer 1 ----
    zero_kernel<<<zero_blocks, TB, 0, stream>>>(agg, NH);
    scatter_add_kernel<<<scatter_blocks, TB, 0, stream>>>(x, src, dst, agg, n_edges);
    conv_gemm_kernel<<<conv_blocks, TB, 0, stream>>>(agg, x, w1rB, w1lB, b1, nullptr, t1, n_nodes);

    // ---- layer 2 ----
    zero_kernel<<<zero_blocks, TB, 0, stream>>>(agg, NH);
    scatter_add_kernel<<<scatter_blocks, TB, 0, stream>>>(t1, src, dst, agg, n_edges);
    conv_gemm_kernel<<<conv_blocks, TB, 0, stream>>>(agg, t1, w2rB, w2lB, b2, nullptr, t2, n_nodes);

    // ---- layer 3 (+ residual of x1) ----
    zero_kernel<<<zero_blocks, TB, 0, stream>>>(agg, NH);
    scatter_add_kernel<<<scatter_blocks, TB, 0, stream>>>(t2, src, dst, agg, n_edges);
    conv_gemm_kernel<<<conv_blocks, TB, 0, stream>>>(agg, t2, w3rB, w3lB, b3, t1, t3, n_nodes);

    // ---- layer 4 (reuses conv3 weights, per reference) ----
    zero_kernel<<<zero_blocks, TB, 0, stream>>>(agg, NH);
    scatter_add_kernel<<<scatter_blocks, TB, 0, stream>>>(t3, src, dst, agg, n_edges);
    conv_gemm_kernel<<<conv_blocks, TB, 0, stream>>>(agg, t3, w3rB, w3lB, b3, nullptr, t4, n_nodes);

    // ---- pooling ----
    const size_t pool_state = (size_t)NGRAPHS * HDIM * 2 + NGRAPHS; // gm + gs + cnt
    zero_kernel<<<(int)((pool_state + TB - 1) / TB), TB, 0, stream>>>(gm, pool_state);
    pool_kernel<<<pool_blocks, TB, 0, stream>>>(t4, batch, (unsigned*)gm, gs, cnt, n_nodes);
    pool_finish_kernel<<<(NGRAPHS * HDIM + TB - 1) / TB, TB, 0, stream>>>((const unsigned*)gm, gs, cnt, h1);

    // ---- MLP head ----
    linear_kernel<<<(NGRAPHS * 128 + TB - 1) / TB, TB, 0, stream>>>(h1, lw1, lb1, h2, NGRAPHS, 256, 128, 1);
    linear_kernel<<<(NGRAPHS * 64 + TB - 1) / TB, TB, 0, stream>>>(h2, lw2, lb2, h3, NGRAPHS, 128, 64, 1);
    linear_kernel<<<(NGRAPHS * NCLASSES + TB - 1) / TB, TB, 0, stream>>>(h3, lw3, lb3, h4, NGRAPHS, 64, NCLASSES, 0);
    logsoftmax_kernel<<<(NGRAPHS + TB - 1) / TB, TB, 0, stream>>>(h4, (float*)d_out, NGRAPHS, NCLASSES);
}
